// BatchResidualLstm_27049704030734
// MI455X (gfx1250) — matrix operand order repaired and checked by reading, not compiled, not run on hardware
//
#include <hip/hip_runtime.h>
#include <math.h>

// ---------------------------------------------------------------------------
// BatchResidualLstm for MI455X (gfx1250, wave32, WMMA)
//   T=1000, N=32, I=H=1024, G=3H=3072
//   Phase 1: BN stats (fixed-order two-pass reduction, float4)
//   Phase 2: normalize -> xn (f32 + f16), weights -> f16 (float4)
//   Phase 3: gx = xn @ W_ii^T + (b_ii+b_ih+b_ic)   [pipelined WMMA f16]
//   Phase 4: 1000 fused step kernels, 512 thr, K split across 2 wave groups,
//            h/c f16 double-buffered across steps, epilogue operands hoisted.
//   K-loops are fully unrolled: SSA renaming kills the cur/next v_mov copies
//   and turns per-iteration address math into immediate offsets.
// ---------------------------------------------------------------------------

#define T_STEPS 1000
#define NB      32             // batch
#define IDIM    1024           // input == hidden
#define HDIM    1024
#define GDIM    3072           // 3*H
#define TN      (T_STEPS * NB) // 32000
#define BN_EPS  1e-5f
#define RB      128            // row blocks for BN reduction (32000/128 = 250)

typedef __attribute__((ext_vector_type(16))) _Float16 v16h;
typedef __attribute__((ext_vector_type(8)))  _Float16 v8h;
typedef __attribute__((ext_vector_type(4)))  _Float16 v4h;
typedef __attribute__((ext_vector_type(8)))  float    v8f;
typedef __attribute__((ext_vector_type(4)))  float    v4f;

// ---------------- WMMA fragment helpers (contiguous K order, k = 16h + e) ----------------
// A (16x32 f16): lane l -> row M = mbase + (l&15); element e holds
// K = kbase + (l>=16?16:0) + e (matches load_b16).
__device__ __forceinline__ v16h load_a16(const _Float16* __restrict__ base,
                                         int lda, int mbase, int kbase, int lane) {
    const int m  = mbase + (lane & 15);
    const int kh = (lane >> 4) << 4;                 // 0 or 16
    const _Float16* p = base + (size_t)m * lda + kbase + kh;
    v8h lo = *(const v8h*)(p);                       // K = kbase+kh .. +7
    v8h hi = *(const v8h*)(p + 8);                   // K = kbase+kh+8 .. +15
    return __builtin_shufflevector(lo, hi, 0,1,2,3,4,5,6,7,8,9,10,11,12,13,14,15);
}

// B (32x16 f16) for C = A * W^T, W row-major [ncols][K]:
// lane l -> col N = nbase + (l&15); element e holds K = kbase + (l>=16?16:0) + e
// => one contiguous 32-byte load per lane. No transpose needed.
__device__ __forceinline__ v16h load_b16(const _Float16* __restrict__ w,
                                         int ldw, int nbase, int kbase, int lane) {
    const int nrow = nbase + (lane & 15);
    const int k    = kbase + ((lane >> 4) << 4);     // 0 or 16
    return *(const v16h*)(w + (size_t)nrow * ldw + k);
}

__device__ __forceinline__ v8f wmma_f16(v16h a, v16h b, v8f c) {
    return __builtin_amdgcn_wmma_f32_16x16x32_f16(false, a, false, b,
                                                  (short)0, c, false, false);
}

__device__ __forceinline__ float sigmf(float x) {
    return 1.0f / (1.0f + __expf(-x));
}

// ---------------------------- Phase 1: BN stats -----------------------------
__global__ void __launch_bounds__(256)
bn_partial_kernel(const float* __restrict__ x,
                  float* __restrict__ psum, float* __restrict__ pss) {
    const int c4 = threadIdx.x * 4;                   // 0..1020 (256 thr * 4 cols)
    const int rb = blockIdx.x;                        // 0..RB-1
    const int r0 = rb * (TN / RB);
    v4f s  = {0.f, 0.f, 0.f, 0.f};
    v4f ss = {0.f, 0.f, 0.f, 0.f};
    for (int r = r0; r < r0 + (TN / RB); ++r) {
        v4f v = *(const v4f*)(x + (size_t)r * IDIM + c4);
        s  += v;
        ss += v * v;
    }
    *(v4f*)(psum + rb * IDIM + c4) = s;
    *(v4f*)(pss  + rb * IDIM + c4) = ss;
}

__global__ void __launch_bounds__(256)
bn_finalize_kernel(const float* __restrict__ psum, const float* __restrict__ pss,
                   const float* __restrict__ gamma, const float* __restrict__ beta,
                   float* __restrict__ scale, float* __restrict__ shift) {
    const int col = blockIdx.x * 256 + threadIdx.x;
    float s = 0.f, ss = 0.f;
    for (int rb = 0; rb < RB; ++rb) {                 // fixed-order: deterministic
        s  += psum[rb * IDIM + col];
        ss += pss [rb * IDIM + col];
    }
    const float inv  = 1.0f / (float)TN;
    const float mean = s * inv;
    const float var  = ss * inv - mean * mean;        // biased, matches torch BN
    const float sc   = gamma[col] * rsqrtf(var + BN_EPS);
    scale[col] = sc;
    shift[col] = beta[col] - mean * sc;
}

// -------------------- Phase 2: conversions / preparation --------------------
__global__ void __launch_bounds__(256)
cvt_f32_to_f16_kernel(const float* __restrict__ src, _Float16* __restrict__ dst,
                      int n4) {                       // n4 = n/4, n always %4==0
    const int i = blockIdx.x * 256 + threadIdx.x;
    if (i < n4) {
        v4f v = *(const v4f*)(src + (size_t)i * 4);
        v4h h;
#pragma unroll
        for (int c = 0; c < 4; ++c) h[c] = (_Float16)v[c];
        *(v4h*)(dst + (size_t)i * 4) = h;
    }
}

__global__ void __launch_bounds__(256)
prep_kernel(const float* __restrict__ bii, const float* __restrict__ bih,
            const float* __restrict__ bic, float* __restrict__ b3,
            _Float16* __restrict__ h16, _Float16* __restrict__ c16,
            float* __restrict__ c32) {
    const int i = blockIdx.x * 256 + threadIdx.x;     // up to 2*NB*HDIM
    if (i < GDIM)      b3[i]  = bii[i] + bih[i] + bic[i];
    if (i < NB * HDIM) c32[i] = 0.f;
    if (i < 2 * NB * HDIM) {                          // both double-buffers
        h16[i] = (_Float16)0.f;
        c16[i] = (_Float16)0.f;
    }
}

__global__ void __launch_bounds__(256)
normalize_kernel(const float* __restrict__ x, const float* __restrict__ scale,
                 const float* __restrict__ shift,
                 float* __restrict__ xnf, _Float16* __restrict__ xnh) {
    const size_t i4 = ((size_t)blockIdx.x * 256 + threadIdx.x) * 4;  // TN*IDIM exact
    const int col   = (int)(i4 & (IDIM - 1));
    v4f xv = *(const v4f*)(x + i4);
    v4f sc = *(const v4f*)(scale + col);
    v4f sh = *(const v4f*)(shift + col);
    v4f v  = xv * sc + sh;
    *(v4f*)(xnf + i4) = v;
    v4h h;
#pragma unroll
    for (int c = 0; c < 4; ++c) h[c] = (_Float16)v[c];
    *(v4h*)(xnh + i4) = h;
}

// ------------- Phase 3: gx = xn @ W_ii^T + b3  (big WMMA GEMM) --------------
// Block: 256 threads = 8 waves. Block tile M=64, N=128.
// Wave (mw = wave&3, nh = wave>>2): 16(M) x 64(N), 4 accumulators.
// Fully-unrolled 2-deep software pipeline: loads for k+64/k+96 are issued
// right after the WMMAs on k/k+32; unrolling makes the stage rotation pure SSA
// renaming (no v_mov copies) and folds k into immediate load offsets.
__global__ void __launch_bounds__(256)
gemm_gx_kernel(const _Float16* __restrict__ A,   // [TN, IDIM]
               const _Float16* __restrict__ W,   // [GDIM, IDIM] row-major
               const float*   __restrict__ b3,   // [GDIM]
               float*         __restrict__ C) {  // [TN, GDIM]
    const int lane  = threadIdx.x & 31;
    const int wave  = threadIdx.x >> 5;
    const int mw    = wave & 3;
    const int nh    = wave >> 2;
    const int mbase = blockIdx.y * 64 + mw * 16;
    const int nbase = blockIdx.x * 128 + nh * 64;

    v8f acc[4] = {};
    v16h a0 = load_a16(A, IDIM, mbase, 0, lane);
    v16h a1 = load_a16(A, IDIM, mbase, 32, lane);
    v16h b0[4], b1[4];
#pragma unroll
    for (int nt = 0; nt < 4; ++nt) {
        b0[nt] = load_b16(W, IDIM, nbase + nt * 16, 0, lane);
        b1[nt] = load_b16(W, IDIM, nbase + nt * 16, 32, lane);
    }
#pragma unroll
    for (int kb = 0; kb + 64 < IDIM; kb += 64) {      // 15 iterations, kb = 0..896
#pragma unroll
        for (int nt = 0; nt < 4; ++nt)
            acc[nt] = wmma_f16(a0, b0[nt], acc[nt]);
        a0 = load_a16(A, IDIM, mbase, kb + 64, lane);
#pragma unroll
        for (int nt = 0; nt < 4; ++nt)
            b0[nt] = load_b16(W, IDIM, nbase + nt * 16, kb + 64, lane);
#pragma unroll
        for (int nt = 0; nt < 4; ++nt)
            acc[nt] = wmma_f16(a1, b1[nt], acc[nt]);
        a1 = load_a16(A, IDIM, mbase, kb + 96, lane);
#pragma unroll
        for (int nt = 0; nt < 4; ++nt)
            b1[nt] = load_b16(W, IDIM, nbase + nt * 16, kb + 96, lane);
    }
#pragma unroll
    for (int nt = 0; nt < 4; ++nt) {                  // final k = 960, 992
        acc[nt] = wmma_f16(a0, b0[nt], acc[nt]);
        acc[nt] = wmma_f16(a1, b1[nt], acc[nt]);
    }
    const int mrow = mbase + ((lane >> 4) << 3);
#pragma unroll
    for (int nt = 0; nt < 4; ++nt) {
        const int col  = nbase + nt * 16 + (lane & 15);
        const float bb = b3[col];
#pragma unroll
        for (int r = 0; r < 8; ++r)
            C[(size_t)(mrow + r) * GDIM + col] = acc[nt][r] + bb;
    }
}

// --------------------- Phase 4: fused recurrent step ------------------------
// Grid: HDIM/32 = 32 blocks x 512 threads; block owns H-columns [c0, c0+32).
// 16 waves: wave -> (kh = K-half, s = slice {i,f,o,cell}, mh = M-half).
// Each wave reduces K in [kh*512, kh*512+512); the two partials are summed
// from LDS in the pointwise phase. All WMMA branches are wave-uniform.
__global__ void __launch_bounds__(512)
step_kernel(int t,
            const _Float16* __restrict__ hA,  const _Float16* __restrict__ cA,
            _Float16* __restrict__ hW,        _Float16* __restrict__ cW,
            float* __restrict__ c32,
            const _Float16* __restrict__ Wih, const _Float16* __restrict__ Wic,
            const _Float16* __restrict__ Whh,
            const float* __restrict__ gx,     const float* __restrict__ bhh,
            const float* __restrict__ xnf,    float* __restrict__ out) {
    __shared__ float lds[2][4][NB][32];               // 32 KB of 320 KB WGP LDS
    const int tid  = threadIdx.x;
    const int lane = tid & 31;
    const int wave = tid >> 5;                        // 0..15
    const int kh   = wave >> 3;                       // K half
    const int s    = (wave >> 1) & 3;                 // slice
    const int mh   = wave & 1;                        // M half
    const int c0   = blockIdx.x * 32;
    const int k0   = kh * (HDIM / 2);

    // Hoisted pointwise operands (prev-step state; read-only until the barrier)
    float cold[2], xnv[2];
#pragma unroll
    for (int e = 0; e < 2; ++e) {
        const int id = tid + 512 * e;                 // 0..1023
        const int n = id >> 5, j = c0 + (id & 31);
        cold[e] = c32[n * HDIM + j];
        xnv[e]  = xnf[((size_t)t * NB + n) * IDIM + j];
    }

    const int nr = mh * 16 + ((lane >> 4) << 3);      // C-frag row base
    const int nc = lane & 15;                         // C-frag col within tile

    v8f acc[2] = {};
    if (s < 3) {                                      // i / f / o gate slices
        float gxv[2][8] = {};                         // kh==0 wave folds gx in
        if (kh == 0) {
#pragma unroll
            for (int nt = 0; nt < 2; ++nt)
#pragma unroll
                for (int r = 0; r < 8; ++r)
                    gxv[nt][r] = gx[((size_t)t * NB + nr + r) * GDIM
                                    + s * HDIM + c0 + nt * 16 + nc];
        }
        const int wb = s * HDIM + c0;
        v16h ah0 = load_a16(hA, HDIM, mh * 16, k0, lane);
        v16h ac0 = load_a16(cA, HDIM, mh * 16, k0, lane);
        v16h ah1 = load_a16(hA, HDIM, mh * 16, k0 + 32, lane);
        v16h ac1 = load_a16(cA, HDIM, mh * 16, k0 + 32, lane);
        v16h bi0[2], bc0[2], bi1[2], bc1[2];
#pragma unroll
        for (int nt = 0; nt < 2; ++nt) {
            bi0[nt] = load_b16(Wih, HDIM, wb + nt * 16, k0, lane);
            bc0[nt] = load_b16(Wic, HDIM, wb + nt * 16, k0, lane);
            bi1[nt] = load_b16(Wih, HDIM, wb + nt * 16, k0 + 32, lane);
            bc1[nt] = load_b16(Wic, HDIM, wb + nt * 16, k0 + 32, lane);
        }
#pragma unroll
        for (int ki = 0; ki < 7; ++ki) {              // kb = k0 .. k0+384
            const int kb = k0 + ki * 64;
#pragma unroll
            for (int nt = 0; nt < 2; ++nt) {
                acc[nt] = wmma_f16(ah0, bi0[nt], acc[nt]);
                acc[nt] = wmma_f16(ac0, bc0[nt], acc[nt]);
            }
            ah0 = load_a16(hA, HDIM, mh * 16, kb + 64, lane);
            ac0 = load_a16(cA, HDIM, mh * 16, kb + 64, lane);
#pragma unroll
            for (int nt = 0; nt < 2; ++nt) {
                bi0[nt] = load_b16(Wih, HDIM, wb + nt * 16, kb + 64, lane);
                bc0[nt] = load_b16(Wic, HDIM, wb + nt * 16, kb + 64, lane);
            }
#pragma unroll
            for (int nt = 0; nt < 2; ++nt) {
                acc[nt] = wmma_f16(ah1, bi1[nt], acc[nt]);
                acc[nt] = wmma_f16(ac1, bc1[nt], acc[nt]);
            }
            ah1 = load_a16(hA, HDIM, mh * 16, kb + 96, lane);
            ac1 = load_a16(cA, HDIM, mh * 16, kb + 96, lane);
#pragma unroll
            for (int nt = 0; nt < 2; ++nt) {
                bi1[nt] = load_b16(Wih, HDIM, wb + nt * 16, kb + 96, lane);
                bc1[nt] = load_b16(Wic, HDIM, wb + nt * 16, kb + 96, lane);
            }
        }
#pragma unroll
        for (int nt = 0; nt < 2; ++nt) {              // final k = k0+448, k0+480
            acc[nt] = wmma_f16(ah0, bi0[nt], acc[nt]);
            acc[nt] = wmma_f16(ac0, bc0[nt], acc[nt]);
            acc[nt] = wmma_f16(ah1, bi1[nt], acc[nt]);
            acc[nt] = wmma_f16(ac1, bc1[nt], acc[nt]);
        }
#pragma unroll
        for (int nt = 0; nt < 2; ++nt)
#pragma unroll
            for (int r = 0; r < 8; ++r)
                lds[kh][s][nr + r][nt * 16 + nc] = acc[nt][r] + gxv[nt][r];
    } else {                                          // cell pre-activation slice
        float bb[2] = {};
        if (kh == 0) {
#pragma unroll
            for (int nt = 0; nt < 2; ++nt) bb[nt] = bhh[c0 + nt * 16 + nc];
        }
        v16h ah0 = load_a16(hA, HDIM, mh * 16, k0, lane);
        v16h ah1 = load_a16(hA, HDIM, mh * 16, k0 + 32, lane);
        v16h b0[2], b1[2];
#pragma unroll
        for (int nt = 0; nt < 2; ++nt) {
            b0[nt] = load_b16(Whh, HDIM, c0 + nt * 16, k0, lane);
            b1[nt] = load_b16(Whh, HDIM, c0 + nt * 16, k0 + 32, lane);
        }
#pragma unroll
        for (int ki = 0; ki < 7; ++ki) {              // kb = k0 .. k0+384
            const int kb = k0 + ki * 64;
#pragma unroll
            for (int nt = 0; nt < 2; ++nt)
                acc[nt] = wmma_f16(ah0, b0[nt], acc[nt]);
            ah0 = load_a16(hA, HDIM, mh * 16, kb + 64, lane);
#pragma unroll
            for (int nt = 0; nt < 2; ++nt)
                b0[nt] = load_b16(Whh, HDIM, c0 + nt * 16, kb + 64, lane);
#pragma unroll
            for (int nt = 0; nt < 2; ++nt)
                acc[nt] = wmma_f16(ah1, b1[nt], acc[nt]);
            ah1 = load_a16(hA, HDIM, mh * 16, kb + 96, lane);
#pragma unroll
            for (int nt = 0; nt < 2; ++nt)
                b1[nt] = load_b16(Whh, HDIM, c0 + nt * 16, kb + 96, lane);
        }
#pragma unroll
        for (int nt = 0; nt < 2; ++nt) {
            acc[nt] = wmma_f16(ah0, b0[nt], acc[nt]);
            acc[nt] = wmma_f16(ah1, b1[nt], acc[nt]);
        }
#pragma unroll
        for (int nt = 0; nt < 2; ++nt)
#pragma unroll
            for (int r = 0; r < 8; ++r)
                lds[kh][3][nr + r][nt * 16 + nc] = acc[nt][r] + bb[nt];
    }
    __syncthreads();

#pragma unroll
    for (int e = 0; e < 2; ++e) {                     // 32x32 elems / 512 threads
        const int id = tid + 512 * e;
        const int n  = id >> 5;
        const int jl = id & 31;
        const int j  = c0 + jl;
        const float ig   = sigmf(lds[0][0][n][jl] + lds[1][0][n][jl]);
        const float fg   = sigmf(lds[0][1][n][jl] + lds[1][1][n][jl]);
        const float og   = sigmf(lds[0][2][n][jl] + lds[1][2][n][jl]);
        const float cell = tanhf(lds[0][3][n][jl] + lds[1][3][n][jl]);
        const float cy   = fg * cold[e] + ig * cell;
        const float hy   = og * (tanhf(cy) + xnv[e]);
        c32[n * HDIM + j] = cy;                       // fp32 master state
        cW [n * HDIM + j] = (_Float16)cy;             // f16 GEMM inputs (dbl-buffered)
        hW [n * HDIM + j] = (_Float16)hy;
        out[((size_t)t * NB + n) * HDIM + j] = hy;    // ys[t]
    }
}

// ------------------------------- launcher -----------------------------------
extern "C" void kernel_launch(void* const* d_in, const int* in_sizes, int n_in,
                              void* d_out, int out_size, void* d_ws, size_t ws_size,
                              hipStream_t stream) {
    const float* x     = (const float*)d_in[0];
    const float* gamma = (const float*)d_in[1];
    const float* beta  = (const float*)d_in[2];
    const float* W_ii  = (const float*)d_in[3];
    const float* W_ih  = (const float*)d_in[4];
    const float* W_ic  = (const float*)d_in[5];
    const float* b_ii  = (const float*)d_in[6];
    const float* b_ih  = (const float*)d_in[7];
    const float* b_ic  = (const float*)d_in[8];
    const float* W_hh  = (const float*)d_in[9];
    const float* b_hh  = (const float*)d_in[10];
    // d_in[11] = output_lengths: unused (reference returns full ys)
    float* out = (float*)d_out;

    // Workspace carve (256-B aligned chunks); total ~612 MB.
    char* p = (char*)d_ws;
    auto carve = [&](size_t bytes) -> char* {
        char* r = p;
        p += (bytes + 255) & ~(size_t)255;
        return r;
    };
    float*    psum  = (float*)   carve((size_t)RB * IDIM * 4);
    float*    pss   = (float*)   carve((size_t)RB * IDIM * 4);
    float*    scale = (float*)   carve((size_t)IDIM * 4);
    float*    shift = (float*)   carve((size_t)IDIM * 4);
    float*    b3    = (float*)   carve((size_t)GDIM * 4);
    float*    c32   = (float*)   carve((size_t)NB * HDIM * 4);
    _Float16* h16   = (_Float16*)carve((size_t)2 * NB * HDIM * 2);  // double buffer
    _Float16* c16   = (_Float16*)carve((size_t)2 * NB * HDIM * 2);  // double buffer
    _Float16* Wii16 = (_Float16*)carve((size_t)GDIM * IDIM * 2);
    _Float16* Wih16 = (_Float16*)carve((size_t)GDIM * HDIM * 2);
    _Float16* Wic16 = (_Float16*)carve((size_t)GDIM * HDIM * 2);
    _Float16* Whh16 = (_Float16*)carve((size_t)HDIM * HDIM * 2);
    _Float16* xnh   = (_Float16*)carve((size_t)TN * IDIM * 2);
    float*    xnf   = (float*)   carve((size_t)TN * IDIM * 4);
    float*    gx    = (float*)   carve((size_t)TN * GDIM * 4);

    // Phase 1: BN statistics
    bn_partial_kernel <<<dim3(RB),         256, 0, stream>>>(x, psum, pss);
    bn_finalize_kernel<<<dim3(IDIM / 256), 256, 0, stream>>>(psum, pss, gamma, beta,
                                                             scale, shift);

    // Phase 2: weight conversion, bias fold, state init, normalization
    {
        const int nw4 = (GDIM * IDIM) / 4;   // 786,432
        cvt_f32_to_f16_kernel<<<(nw4 + 255) / 256, 256, 0, stream>>>(W_ii, Wii16, nw4);
        cvt_f32_to_f16_kernel<<<(nw4 + 255) / 256, 256, 0, stream>>>(W_ih, Wih16, nw4);
        cvt_f32_to_f16_kernel<<<(nw4 + 255) / 256, 256, 0, stream>>>(W_ic, Wic16, nw4);
        const int nh4 = (HDIM * HDIM) / 4;   // 262,144
        cvt_f32_to_f16_kernel<<<(nh4 + 255) / 256, 256, 0, stream>>>(W_hh, Whh16, nh4);
    }
    prep_kernel<<<(2 * NB * HDIM) / 256, 256, 0, stream>>>(b_ii, b_ih, b_ic, b3,
                                                           h16, c16, c32);
    normalize_kernel<<<((size_t)TN * IDIM / 4) / 256, 256, 0, stream>>>(x, scale, shift,
                                                                        xnf, xnh);

    // Phase 3: gx = xn @ W_ii^T + b3   (grid: 24 x 500 blocks of 64x128)
    gemm_gx_kernel<<<dim3(GDIM / 128, TN / 64), 256, 0, stream>>>(xnh, Wii16, b3, gx);

    // Phase 4: 1000 fused recurrent steps (stream-ordered, double-buffered h/c)
    const size_t HC = (size_t)NB * HDIM;
    for (int t = 0; t < T_STEPS; ++t) {
        const int rd = t & 1, wr = (t + 1) & 1;
        step_kernel<<<HDIM / 32, 512, 0, stream>>>(
            t, h16 + rd * HC, c16 + rd * HC, h16 + wr * HC, c16 + wr * HC,
            c32, Wih16, Wic16, Whh16, gx, b_hh, xnf, out);
    }
}
